// Block_5360119185819
// MI455X (gfx1250) — matrix operand order repaired and checked by reading, not compiled, not run on hardware
//
#include <hip/hip_runtime.h>
#include <math.h>

#define N_SITES 100000
#define DIM     128
#define HID     512
#define KVOL    343

typedef __attribute__((ext_vector_type(16))) _Float16 v16h;
typedef __attribute__((ext_vector_type(8)))  float    v8f;

union HF16 { v16h v; _Float16 h[16]; uint4 q[2]; };

// ---------- WMMA helper (16x16x32 f16 -> f32) ----------
__device__ inline v8f wmma_f16(v16h a, v16h b, v8f c) {
  return __builtin_amdgcn_wmma_f32_16x16x32_f16(
      /*neg_a=*/false, a, /*neg_b=*/false, b,
      /*c_mod=*/(short)0, c, /*reuse_a=*/false, /*reuse_b=*/false);
}

// A fragment: 16x32 f16 tile from LDS (row-major, padded stride).
// Lane l<16: row M=l, K in {kb..kb+15}, kb=kk (+16 for lanes>=16).
// Each half-group is 16B contiguous -> compiler emits 2x ds_load_b128.
__device__ inline v16h load_a_frag(const _Float16* __restrict__ Al, int stride,
                                   int row, int kk, int laneHi) {
  HF16 f;
  const _Float16* base = Al + row * stride;
  int kb = kk + (laneHi ? 16 : 0);
#pragma unroll
  for (int j = 0; j < 4; ++j) {
    f.h[2 * j]         = base[kb + 2 * j];
    f.h[2 * j + 1]     = base[kb + 2 * j + 1];
    f.h[8 + 2 * j]     = base[kb + 8 + 2 * j];
    f.h[8 + 2 * j + 1] = base[kb + 8 + 2 * j + 1];
  }
  return f.v;
}

// B fragment from pre-swizzled weights: lane's 16 halves are contiguous
// (32 bytes) -> two global_load_b128, zero packing ALU.
__device__ inline v16h load_b_packed(const _Float16* __restrict__ Bp,
                                     int frag, int lane) {
  HF16 f;
  const uint4* p = (const uint4*)(Bp + ((size_t)frag * 512) + lane * 16);
  f.q[0] = p[0];
  f.q[1] = p[1];
  return f.v;
}

// ---------- kernel 1: pack w1 -> WMMA-B-fragment-layout f16; zero sumsq ----
// packed index e: frag f = e/512, lane l = (e/16)&31, j2 = e&15
//   kk = (f / (HID/16)) * 32, ct = f % (HID/16)
//   K  = kk + (l>=16 ? 16 : 0) + j2,  col = ct*16 + (l&15)
__global__ __launch_bounds__(256) void prep_kernel(
    const float* __restrict__ w1, _Float16* __restrict__ w1p,
    float* __restrict__ sumsq) {
  int e = blockIdx.x * 256 + threadIdx.x;   // 65536 = DIM*HID
  int f  = e >> 9;
  int l  = (e >> 4) & 31;
  int j2 = e & 15;
  int kk = (f >> 5) << 5;                   // 32 col-tiles for width 512
  int ct = f & 31;
  int K  = kk + ((l & 16) ? 16 : 0) + j2;
  int col = ct * 16 + (l & 15);
  w1p[e] = (_Float16)w1[K * HID + col];
  if (e < HID) sumsq[e] = 0.0f;
}

// ---------- kernel 2: sparse depthwise conv + LayerNorm -> f16 ----------
// one wave per site, 4 channels per lane (float4), shuffle-based LN.
__global__ __launch_bounds__(256) void dwconv_ln_kernel(
    const float* __restrict__ feats, const int* __restrict__ nbr,
    const float* __restrict__ dw_w, const float* __restrict__ dw_b,
    const float* __restrict__ ln_g, const float* __restrict__ ln_b,
    _Float16* __restrict__ xln) {
  int wave = threadIdx.x >> 5, lane = threadIdx.x & 31;
  int site = blockIdx.x * 8 + wave;
  int c = lane * 4;
  const int* __restrict__ row = nbr + (long)site * KVOL;
  float4 acc = make_float4(0.f, 0.f, 0.f, 0.f);
  for (int k = 0; k < KVOL; ++k) {
    int idx = row[k];                       // uniform across wave; sentinel==N
    if (idx < N_SITES) {
      float4 f = *(const float4*)(feats + (long)idx * DIM + c);
      float4 w = *(const float4*)(dw_w + k * DIM + c);
      acc.x += f.x * w.x; acc.y += f.y * w.y;
      acc.z += f.z * w.z; acc.w += f.w * w.w;
    }
  }
  float4 bb = *(const float4*)(dw_b + c);
  acc.x += bb.x; acc.y += bb.y; acc.z += bb.z; acc.w += bb.w;
  float s  = acc.x + acc.y + acc.z + acc.w;
  float sq = acc.x * acc.x + acc.y * acc.y + acc.z * acc.z + acc.w * acc.w;
#pragma unroll
  for (int o = 16; o > 0; o >>= 1) {
    s  += __shfl_xor(s,  o, 32);
    sq += __shfl_xor(sq, o, 32);
  }
  float mu   = s * (1.0f / DIM);
  float var  = sq * (1.0f / DIM) - mu * mu;
  float rstd = rsqrtf(var + 1e-6f);
  float4 g = *(const float4*)(ln_g + c);
  float4 b = *(const float4*)(ln_b + c);
  HF16 o4;   // pack 4 halves, store 8B
  o4.h[0] = (_Float16)((acc.x - mu) * rstd * g.x + b.x);
  o4.h[1] = (_Float16)((acc.y - mu) * rstd * g.y + b.y);
  o4.h[2] = (_Float16)((acc.z - mu) * rstd * g.z + b.z);
  o4.h[3] = (_Float16)((acc.w - mu) * rstd * g.w + b.w);
  *(uint2*)(xln + (long)site * DIM + c) = *(uint2*)&o4.h[0];
}

// ---------- kernel 3: h = gelu(xln @ w1 + b1), f16 out, per-channel sum(h^2) ----------
#define LDSA1 (DIM + 8)
__global__ __launch_bounds__(256) void gemm1_kernel(
    const _Float16* __restrict__ xln, const _Float16* __restrict__ w1p,
    const float* __restrict__ b1, _Float16* __restrict__ hbuf,
    float* __restrict__ sumsq) {
  __shared__ _Float16 Al[16 * LDSA1];
  int t = threadIdx.x;
  int site0 = blockIdx.x * 16;
  { // stage 16x128 f16 A-tile (4 KB), one b128 per thread
    int row = t >> 4, ks = (t & 15) * 8;
    uint4 v = *(const uint4*)(xln + (long)(site0 + row) * DIM + ks);
    *(uint4*)(Al + row * LDSA1 + ks) = v;
  }
  __syncthreads();
  int wave = t >> 5, lane = t & 31;
  int laneHi = lane >> 4, ln15 = lane & 15;
  v8f z = {};
  v8f acc[4] = {z, z, z, z};                 // 8 waves x 4 tiles = 512 cols
#pragma unroll
  for (int kt = 0; kt < 4; ++kt) {
    v16h a = load_a_frag(Al, LDSA1, ln15, kt * 32, laneHi);
#pragma unroll
    for (int tt = 0; tt < 4; ++tt) {
      int ct = wave * 4 + tt;                // col-tile in [0,32)
      v16h b = load_b_packed(w1p, kt * 32 + ct, lane);
      acc[tt] = wmma_f16(a, b, acc[tt]);
    }
  }
#pragma unroll
  for (int tt = 0; tt < 4; ++tt) {
    int colg = wave * 64 + tt * 16 + ln15;
    float bias = b1[colg];
    float ss = 0.f;
#pragma unroll
    for (int r = 0; r < 8; ++r) {
      int m = laneHi * 8 + r;
      float v = acc[tt][r] + bias;
      float gel = 0.5f * v * (1.0f + erff(v * 0.70710678118654752f)); // exact GELU
      hbuf[(long)(site0 + m) * HID + colg] = (_Float16)gel;
      ss += gel * gel;
    }
    ss += __shfl_xor(ss, 16, 32);            // fold rows 8..15 into lanes 0..15
    if (!laneHi) atomicAdd(&sumsq[colg], ss);
  }
}

// ---------- kernel 4: GRN -> pack scaled w2 into B-fragment layout f16,
//                        fold grn_b@w2 + b2 into cbias ----------
__global__ __launch_bounds__(512) void grn_kernel(
    const float* __restrict__ sumsq, const float* __restrict__ grn_g,
    const float* __restrict__ grn_b, const float* __restrict__ w2,
    const float* __restrict__ b2, _Float16* __restrict__ w2p,
    float* __restrict__ cbias) {
  __shared__ float red[HID];
  __shared__ float s_sh[HID];
  int t = threadIdx.x;
  float g = sqrtf(sumsq[t]);
  red[t] = g;
  __syncthreads();
  for (int o = 256; o > 0; o >>= 1) {
    if (t < o) red[t] += red[t + o];
    __syncthreads();
  }
  float mean = red[0] * (1.0f / HID);
  float nx = g / (mean + 1e-6f);
  s_sh[t] = grn_g[t] * nx + 1.0f;            // x*(grn_g*nx + 1) + grn_b
  __syncthreads();
  for (int e = t; e < HID * DIM; e += 512) { // pack scaled w2 (width 128)
    int f  = e >> 9;
    int l  = (e >> 4) & 31;
    int j2 = e & 15;
    int kk = (f >> 3) << 5;                  // 8 col-tiles for width 128
    int ct = f & 7;
    int K  = kk + ((l & 16) ? 16 : 0) + j2;
    int col = ct * 16 + (l & 15);
    w2p[e] = (_Float16)(w2[K * DIM + col] * s_sh[K]);
  }
  if (t < DIM) {
    float a = b2[t];
    for (int c = 0; c < HID; ++c) a += grn_b[c] * w2[c * DIM + t];
    cbias[t] = a;
  }
}

// ---------- kernel 5: out = (h*s) @ w2 + cbias + feats ----------
#define LDSA2 (HID + 8)
__global__ __launch_bounds__(256) void gemm2_kernel(
    const _Float16* __restrict__ hbuf, const _Float16* __restrict__ w2p,
    const float* __restrict__ cbias, const float* __restrict__ feats,
    float* __restrict__ out) {
  __shared__ _Float16 Al[16 * LDSA2];        // 16x512 f16 tile (padded)
  int t = threadIdx.x;
  int site0 = blockIdx.x * 16;
#pragma unroll
  for (int i = 0; i < 4; ++i) {              // 16 KB staged, 4 b128 per thread
    int ci = t + i * 256;
    int row = ci >> 6, ks = (ci & 63) * 8;
    uint4 v = *(const uint4*)(hbuf + (long)(site0 + row) * HID + ks);
    *(uint4*)(Al + row * LDSA2 + ks) = v;
  }
  __syncthreads();
  int wave = t >> 5, lane = t & 31;
  int laneHi = lane >> 4, ln15 = lane & 15;
  int colg = wave * 16 + ln15;               // 8 waves x 16 = 128 cols
  v8f acc = {};
#pragma unroll
  for (int kt = 0; kt < 16; ++kt) {
    v16h a = load_a_frag(Al, LDSA2, ln15, kt * 32, laneHi);
    v16h b = load_b_packed(w2p, kt * 8 + wave, lane);
    acc = wmma_f16(a, b, acc);
  }
  float cb = cbias[colg];
#pragma unroll
  for (int r = 0; r < 8; ++r) {
    int m = laneHi * 8 + r;
    long o = (long)(site0 + m) * DIM + colg;
    out[o] = acc[r] + cb + feats[o];
  }
}

// ---------- workspace layout ----------
#define XLN_BYTES  ((size_t)N_SITES * DIM * 2)       //  25.6 MB  f16
#define H_BYTES    ((size_t)N_SITES * HID * 2)       // 102.4 MB  f16
#define W1P_BYTES  ((size_t)DIM * HID * 2)           // 128 KB
#define W2P_BYTES  ((size_t)HID * DIM * 2)           // 128 KB
#define SS_BYTES   ((size_t)HID * 4)

extern "C" void kernel_launch(void* const* d_in, const int* in_sizes, int n_in,
                              void* d_out, int out_size, void* d_ws, size_t ws_size,
                              hipStream_t stream) {
  const float* feats = (const float*)d_in[0];
  const int*   nbr   = (const int*)d_in[1];
  const float* dw_w  = (const float*)d_in[2];
  const float* dw_b  = (const float*)d_in[3];
  const float* ln_g  = (const float*)d_in[4];
  const float* ln_b  = (const float*)d_in[5];
  const float* w1    = (const float*)d_in[6];
  const float* b1    = (const float*)d_in[7];
  const float* grn_g = (const float*)d_in[8];
  const float* grn_b = (const float*)d_in[9];
  const float* w2    = (const float*)d_in[10];
  const float* b2    = (const float*)d_in[11];
  float* out = (float*)d_out;

  char* ws = (char*)d_ws;
  _Float16* xln   = (_Float16*)(ws);
  _Float16* hbuf  = (_Float16*)(ws + XLN_BYTES);
  _Float16* w1p   = (_Float16*)(ws + XLN_BYTES + H_BYTES);
  _Float16* w2p   = (_Float16*)(ws + XLN_BYTES + H_BYTES + W1P_BYTES);
  float*    sumsq = (float*)   (ws + XLN_BYTES + H_BYTES + W1P_BYTES + W2P_BYTES);
  float*    cbias = (float*)   (ws + XLN_BYTES + H_BYTES + W1P_BYTES + W2P_BYTES + SS_BYTES);

  prep_kernel<<<(DIM * HID) / 256, 256, 0, stream>>>(w1, w1p, sumsq);
  dwconv_ln_kernel<<<N_SITES / 8, 256, 0, stream>>>(feats, nbr, dw_w, dw_b,
                                                    ln_g, ln_b, xln);
  gemm1_kernel<<<N_SITES / 16, 256, 0, stream>>>(xln, w1p, b1, hbuf, sumsq);
  grn_kernel<<<1, 512, 0, stream>>>(sumsq, grn_g, grn_b, w2, b2, w2p, cbias);
  gemm2_kernel<<<N_SITES / 16, 256, 0, stream>>>(hbuf, w2p, cbias, feats, out);
}
